// VarianceAdaptor_24489903522414
// MI455X (gfx1250) — matrix operand order repaired and checked by reading, not compiled, not run on hardware
//
#include <hip/hip_runtime.h>

// Problem constants
#define BB   32
#define SS   256
#define HH   384
#define PHH  384
#define MELM 2048
#define KSZ  1152   // 3 * H (im2col reduction length)
#define BROW 136    // LDS row stride in f16 for a 128-wide K chunk (+8 pad -> bank-conflict-free)

typedef __attribute__((ext_vector_type(16))) _Float16 v16h;
typedef __attribute__((ext_vector_type(8)))  _Float16 v8h;
typedef __attribute__((ext_vector_type(8)))  float    v8f;
typedef __attribute__((ext_vector_type(4)))  int      v4i;

union AFrag { v16h v; v8h h[2]; };

// gfx1250 async global->LDS staging (guarded: falls back to load+ds_store)
#if defined(__AMDGCN__) && __has_builtin(__builtin_amdgcn_global_load_async_to_lds_b128)
#define USE_ASYNC_LDS 1
#else
#define USE_ASYNC_LDS 0
#endif

__device__ __forceinline__ void wait_async_lds() {
#if USE_ASYNC_LDS
#if __has_builtin(__builtin_amdgcn_s_wait_asynccnt)
  __builtin_amdgcn_s_wait_asynccnt(0);
#else
  asm volatile("s_wait_asynccnt 0x0" ::: "memory");
#endif
#endif
}

// ---------------------------------------------------------------------------
// Conv1d(K=3, SAME) + bias + ReLU as a WMMA GEMM.
// 256-thread block (8 waves). Block tile: 128 positions x 64 channels.
// Weight panel staged per-128-K-chunk into double-buffered LDS (async path),
// shared by all 8 waves; each wave computes a 16(pos) x 64(ch) tile.
//   in16 : (B*T, H)   f16 activations
//   wpk  : (PH, 1152) f16 weights packed as [co][tap*H + ci]
//   out32: (B*T, PH)  f32  relu(conv + bias)
// ---------------------------------------------------------------------------
__global__ __launch_bounds__(256) void conv_wmma_k(
    const _Float16* __restrict__ in16, const _Float16* __restrict__ wpk,
    const float* __restrict__ bias, float* __restrict__ out32, int T)
{
  __shared__ _Float16 Bs[2][64 * BROW];   // 2 x 17 KB

  const int tid   = threadIdx.x;
  const int wave  = tid >> 5;
  const int lane  = tid & 31;
  const int half  = lane >> 4;            // K-half selector for A/B fragments
  const int n     = lane & 15;            // row (A load) / col (B) index
  const int nbase = blockIdx.y * 64;

  // Stage one 64(ch) x 128(K) weight chunk into LDS buffer `buf`.
  auto stage = [&](int kc, int buf) {
    // 64 rows x 16 chunks of 8 f16 (16B) = 1024 chunks; 4 per thread.
    for (int i = tid; i < 64 * 16; i += 256) {
      int row = i >> 4;
      int c8  = i & 15;
      const _Float16* src = wpk + (long long)(nbase + row) * KSZ + kc * 128 + c8 * 8;
      _Float16*       dst = &Bs[buf][row * BROW + c8 * 8];
#if USE_ASYNC_LDS
      __builtin_amdgcn_global_load_async_to_lds_b128(
          (__attribute__((address_space(1))) v4i*)src,
          (__attribute__((address_space(3))) v4i*)dst, 0, 0);
#else
      *(v8h*)dst = *(const v8h*)src;
#endif
    }
  };

  // Per-wave M tile
  const int mtile = blockIdx.x * 8 + wave;
  const int r = mtile * 16 + n;           // A-fragment source row for this lane
  const int b = r / T;
  const int t = r - b * T;
  const long long batchBase = (long long)b * T * HH;

  v8f acc0 = {}, acc1 = {}, acc2 = {}, acc3 = {};
  const v8h z8 = {};

  stage(0, 0);
  wait_async_lds();
  __syncthreads();

  const int NKC = KSZ / 128;              // 9 chunks of K=128
  for (int kc = 0; kc < NKC; ++kc) {
    const int cur = kc & 1;
    if (kc + 1 < NKC) stage(kc + 1, cur ^ 1);   // prefetch next chunk

    const _Float16* bs = &Bs[cur][0];
#pragma unroll
    for (int k4 = 0; k4 < 4; ++k4) {
      const int kk = kc * 4 + k4;
      // --- A fragment: two contiguous 16B global loads (zero-pad at edges) ---
      AFrag a;
      {
        int gk  = kk * 32 + half * 16;          // K offsets {0..7}+half*16
        int tap = gk / HH, c = gk - tap * HH;   // chunk never crosses a tap
        int st  = t + tap - 1;
        a.h[0] = (st >= 0 && st < T)
          ? *(const v8h*)(in16 + batchBase + (long long)st * HH + c) : z8;
      }
      {
        int gk  = kk * 32 + half * 16 + 8;      // K offsets {8..15}+half*16
        int tap = gk / HH, c = gk - tap * HH;
        int st  = t + tap - 1;
        a.h[1] = (st >= 0 && st < T)
          ? *(const v8h*)(in16 + batchBase + (long long)st * HH + c) : z8;
      }

      // --- B fragments: one contiguous 32B LDS read per subtile ---
      const _Float16* bp = bs + k4 * 32 + half * 16;
      v16h b0 = *(const v16h*)(bp + ( 0 + n) * BROW);
      v16h b1 = *(const v16h*)(bp + (16 + n) * BROW);
      v16h b2 = *(const v16h*)(bp + (32 + n) * BROW);
      v16h b3 = *(const v16h*)(bp + (48 + n) * BROW);

      acc0 = __builtin_amdgcn_wmma_f32_16x16x32_f16(false, a.v, false, b0, (short)0, acc0, false, false);
      acc1 = __builtin_amdgcn_wmma_f32_16x16x32_f16(false, a.v, false, b1, (short)0, acc1, false, false);
      acc2 = __builtin_amdgcn_wmma_f32_16x16x32_f16(false, a.v, false, b2, (short)0, acc2, false, false);
      acc3 = __builtin_amdgcn_wmma_f32_16x16x32_f16(false, a.v, false, b3, (short)0, acc3, false, false);
    }

    wait_async_lds();     // next chunk landed (this wave's async ops)
    __syncthreads();      // all waves done with `cur`, all staging visible
  }

  // Epilogue: C/D layout — VGPR i holds M = i + half*8, N = lane&15
  const float bb0 = bias[nbase + n];
  const float bb1 = bias[nbase + 16 + n];
  const float bb2 = bias[nbase + 32 + n];
  const float bb3 = bias[nbase + 48 + n];
  for (int i = 0; i < 8; ++i) {
    int pr = mtile * 16 + i + half * 8;
    long long o = (long long)pr * PHH + nbase + n;
    out32[o]      = fmaxf(acc0[i] + bb0, 0.f);
    out32[o + 16] = fmaxf(acc1[i] + bb1, 0.f);
    out32[o + 32] = fmaxf(acc2[i] + bb2, 0.f);
    out32[o + 48] = fmaxf(acc3[i] + bb3, 0.f);
  }
}

// ---------------------------------------------------------------------------
// LayerNorm over H=384 channels; f32 in, f16 out (next GEMM input).
// One block (128 threads, 3 channels/thread) per position.
// ---------------------------------------------------------------------------
__global__ __launch_bounds__(128) void ln_k(const float* __restrict__ in,
    const float* __restrict__ g, const float* __restrict__ be,
    _Float16* __restrict__ out)
{
  const int p   = blockIdx.x;
  const int tid = threadIdx.x;
  const long long base = (long long)p * HH;
  float v0 = in[base + tid];
  float v1 = in[base + tid + 128];
  float v2 = in[base + tid + 256];

  __shared__ float red[128];
  red[tid] = v0 + v1 + v2;
  __syncthreads();
  for (int off = 64; off > 0; off >>= 1) {
    if (tid < off) red[tid] += red[tid + off];
    __syncthreads();
  }
  float mean = red[0] * (1.f / HH);
  __syncthreads();

  float d0 = v0 - mean, d1 = v1 - mean, d2 = v2 - mean;
  red[tid] = d0 * d0 + d1 * d1 + d2 * d2;
  __syncthreads();
  for (int off = 64; off > 0; off >>= 1) {
    if (tid < off) red[tid] += red[tid + off];
    __syncthreads();
  }
  float rs = rsqrtf(red[0] * (1.f / HH) + 1e-5f);

  out[base + tid]       = (_Float16)(d0 * rs * g[tid]       + be[tid]);
  out[base + tid + 128] = (_Float16)(d1 * rs * g[tid + 128] + be[tid + 128]);
  out[base + tid + 256] = (_Float16)(d2 * rs * g[tid + 256] + be[tid + 256]);
}

// ---------------------------------------------------------------------------
// Predictor head: per-position dot(h[384], lw) + lb, optionally masked.
// One wave per position (4 waves/block), wave32 shuffle reduction.
// ---------------------------------------------------------------------------
__global__ __launch_bounds__(128) void head_k(const _Float16* __restrict__ h16,
    const float* __restrict__ lw, const float* __restrict__ lb,
    const float* __restrict__ mask, float* __restrict__ out, int npos)
{
  const int p    = blockIdx.x * 4 + (threadIdx.x >> 5);
  const int lane = threadIdx.x & 31;
  if (p >= npos) return;
  const _Float16* row = h16 + (long long)p * PHH;
  float s = 0.f;
  for (int c = lane; c < PHH; c += 32) s += (float)row[c] * lw[c];
  for (int off = 16; off > 0; off >>= 1) s += __shfl_xor(s, off, 32);
  if (lane == 0) {
    float v = s + lb[0];
    if (mask) v *= mask[p];
    out[p] = v;
  }
}

// ---------------------------------------------------------------------------
// Length regulation: dur = max(gt,1); cumsum; searchsorted(right); gather x
// into expanded (f32 + f16), write mel_mask (0/1 float) and total (float).
// One block per batch.
// ---------------------------------------------------------------------------
__global__ __launch_bounds__(256) void expand_k(const float* __restrict__ x,
    const int* __restrict__ gt, float* __restrict__ exp32,
    _Float16* __restrict__ exp16, float* __restrict__ out_mask,
    float* __restrict__ out_total)
{
  const int b   = blockIdx.x;
  const int tid = threadIdx.x;
  __shared__ int   end_s[SS];
  __shared__ short idx_s[MELM];
  __shared__ int   total_sh;

  if (tid == 0) {
    int c = 0;
    for (int s = 0; s < SS; ++s) {
      int d = gt[b * SS + s];
      if (d < 1) d = 1;
      c += d;
      end_s[s] = c;
    }
    total_sh = c;
  }
  __syncthreads();
  const int total = total_sh;
  if (tid == 0) out_total[b] = (float)total;

  for (int t = tid; t < MELM; t += 256) {
    int lo = 0, hi = SS;                       // searchsorted side='right'
    while (lo < hi) {
      int mid = (lo + hi) >> 1;
      if (end_s[mid] <= t) lo = mid + 1; else hi = mid;
    }
    if (lo > SS - 1) lo = SS - 1;
    idx_s[t] = (short)lo;
    out_mask[b * MELM + t] = (t < total) ? 1.f : 0.f;
  }
  __syncthreads();

  const long long xb = (long long)b * SS * HH;
  const long long ob = (long long)b * MELM * HH;
  for (long long i = tid; i < (long long)MELM * HH; i += 256) {
    int t = (int)(i / HH);
    int c = (int)(i - (long long)t * HH);
    float v = (t < total) ? x[xb + (long long)idx_s[t] * HH + c] : 0.f;
    exp32[ob + i] = v;
    exp16[ob + i] = (_Float16)v;
  }
}

// ---------------------------------------------------------------------------
// Pack conv weights (PH,H,3) f32 -> (PH, 3*H) f16 im2col layout [co][tap*H+ci]
// ---------------------------------------------------------------------------
__global__ __launch_bounds__(256) void packw_k(const float* __restrict__ w,
                                               _Float16* __restrict__ o)
{
  int i   = blockIdx.x * 256 + threadIdx.x;    // over PH*KSZ, exact grid
  int co  = i / KSZ;
  int rem = i - co * KSZ;
  int tap = rem / HH;
  int ci  = rem - tap * HH;
  o[i] = (_Float16)w[(long long)co * KSZ + ci * 3 + tap];
}

__global__ __launch_bounds__(256) void cvt16_k(const float* __restrict__ a,
                                               _Float16* __restrict__ o, int n)
{
  int i = blockIdx.x * 256 + threadIdx.x;
  if (i < n) o[i] = (_Float16)a[i];
}

// ---------------------------------------------------------------------------
// out = mask ? expanded + pitch*pew + peb + energy*eew + eeb : 0
// ---------------------------------------------------------------------------
__global__ __launch_bounds__(256) void combine_k(const float* __restrict__ exp32,
    const float* __restrict__ pv, const float* __restrict__ ev,
    const float* __restrict__ mask,
    const float* __restrict__ pew, const float* __restrict__ peb,
    const float* __restrict__ eew, const float* __restrict__ eeb,
    float* __restrict__ out)
{
  long long i = (long long)blockIdx.x * 256 + threadIdx.x;
  int pos = (int)(i / HH);
  int c   = (int)(i - (long long)pos * HH);
  float mm = mask[pos];
  float v = exp32[i] + pv[pos] * pew[c] + peb[c] + ev[pos] * eew[c] + eeb[c];
  out[i] = (mm != 0.f) ? v : 0.f;
}

// ---------------------------------------------------------------------------
extern "C" void kernel_launch(void* const* d_in, const int* in_sizes, int n_in,
                              void* d_out, int out_size, void* d_ws, size_t ws_size,
                              hipStream_t stream)
{
  (void)in_sizes; (void)n_in; (void)out_size; (void)ws_size;

  const float* x  = (const float*)d_in[0];
  const int*   gt = (const int*)d_in[2];      // d_in[1] = src_mask (all true)

  auto P = [&](int i) { return (const float*)d_in[i]; };
  struct Pred { const float *b1,*b2,*be1,*be2,*g1,*g2,*lb,*lw,*w1,*w2; };
  auto mk = [&](int base) {
    Pred p; p.b1 = P(base); p.b2 = P(base+1); p.be1 = P(base+2); p.be2 = P(base+3);
    p.g1 = P(base+4); p.g2 = P(base+5); p.lb = P(base+6); p.lw = P(base+7);
    p.w1 = P(base+8); p.w2 = P(base+9); return p;
  };
  // params flattened in sorted-key order: dur, energy, energy_emb_b/w, pitch, pitch_emb_b/w
  Pred pd = mk(3), pe = mk(13), pp = mk(25);
  const float* eeb = P(23); const float* eew = P(24);
  const float* peb = P(35); const float* pew = P(36);

  // workspace carve-out
  char* wptr = (char*)d_ws;
  auto carve = [&](size_t bytes) {
    void* r = wptr; wptr += (bytes + 255) & ~((size_t)255); return r;
  };
  _Float16* x16    = (_Float16*)carve((size_t)BB * SS   * HH  * 2);
  _Float16* exp16  = (_Float16*)carve((size_t)BB * MELM * HH  * 2);
  float*    exp32  = (float*)   carve((size_t)BB * MELM * HH  * 4);
  float*    conv32 = (float*)   carve((size_t)BB * MELM * PHH * 4);
  _Float16* h16    = (_Float16*)carve((size_t)BB * MELM * PHH * 2);
  _Float16* wp[6];
  for (int i = 0; i < 6; ++i) wp[i] = (_Float16*)carve((size_t)PHH * KSZ * 2);

  // output regions (concatenated, return order; ints/bools stored as float)
  float* out_main   = (float*)d_out;                       // (B, MEL, H)
  float* out_ldur   = out_main  + (size_t)BB * MELM * HH;  // (B, S, 1)
  float* out_pitch  = out_ldur  + (size_t)BB * SS;         // (B, MEL, 1)
  float* out_energy = out_pitch + (size_t)BB * MELM;       // (B, MEL, 1)
  float* out_total  = out_energy+ (size_t)BB * MELM;       // (B,)
  float* out_mask   = out_total + BB;                      // (B, MEL)

  const int NW = (PHH * KSZ) / 256;  // 1728
  packw_k<<<NW, 256, 0, stream>>>(pd.w1, wp[0]);
  packw_k<<<NW, 256, 0, stream>>>(pd.w2, wp[1]);
  packw_k<<<NW, 256, 0, stream>>>(pp.w1, wp[2]);
  packw_k<<<NW, 256, 0, stream>>>(pp.w2, wp[3]);
  packw_k<<<NW, 256, 0, stream>>>(pe.w1, wp[4]);
  packw_k<<<NW, 256, 0, stream>>>(pe.w2, wp[5]);

  cvt16_k<<<(BB * SS * HH) / 256, 256, 0, stream>>>(x, x16, BB * SS * HH);
  expand_k<<<BB, 256, 0, stream>>>(x, gt, exp32, exp16, out_mask, out_total);

  auto run_pred = [&](const _Float16* in16, int T, const Pred& p,
                      _Float16* w1p, _Float16* w2p, float* headOut,
                      const float* mask) {
    const int rows = BB * T;
    dim3 g(rows / 128, PHH / 64);       // 8 M-tiles per block, 64-ch panels
    conv_wmma_k<<<g, 256, 0, stream>>>(in16, w1p, p.b1, conv32, T);
    ln_k<<<rows, 128, 0, stream>>>(conv32, p.g1, p.be1, h16);
    conv_wmma_k<<<g, 256, 0, stream>>>(h16, w2p, p.b2, conv32, T);
    ln_k<<<rows, 128, 0, stream>>>(conv32, p.g2, p.be2, h16);
    head_k<<<rows / 4, 128, 0, stream>>>(h16, p.lw, p.lb, mask, headOut, rows);
  };

  run_pred(x16,   SS,   pd, wp[0], wp[1], out_ldur,   nullptr);
  run_pred(exp16, MELM, pp, wp[2], wp[3], out_pitch,  out_mask);
  run_pred(exp16, MELM, pe, wp[4], wp[5], out_energy, out_mask);

  combine_k<<<(BB * MELM * HH) / 256, 256, 0, stream>>>(
      exp32, out_pitch, out_energy, out_mask, pew, peb, eew, eeb, out_main);
}
